// SparseAutoencoder_33638183863055
// MI455X (gfx1250) — matrix operand order repaired and checked by reading, not compiled, not run on hardware
//
#include <hip/hip_runtime.h>
#include <hip/hip_bf16.h>

#define D_MODEL 2048
#define DICT    32768
#define TOPK    32
#define NROWS   2048

typedef __attribute__((ext_vector_type(16))) __bf16 v16bf;
typedef __attribute__((ext_vector_type(8)))  __bf16 v8bf;
typedef __attribute__((ext_vector_type(4)))  __bf16 v4bf;
typedef __attribute__((ext_vector_type(8)))  float  v8f;

// ---------------------------------------------------------------------------
// CDNA5 async copy: global -> LDS, 16 bytes, tracked by ASYNCcnt.
// VDST = VGPR holding LDS byte address, ADDR = VGPR pair with global address.
// ---------------------------------------------------------------------------
__device__ __forceinline__ void async_b128(unsigned lds_addr, const void* gaddr) {
    asm volatile("global_load_async_to_lds_b128 %0, %1, off"
                 :: "v"(lds_addr), "v"(gaddr)
                 : "memory");
}
__device__ __forceinline__ void wait_async0() {
    asm volatile("s_wait_asynccnt 0x0" ::: "memory");
}

// ---------------------------------------------------------------------------
// f32 -> bf16 conversion (RNE via clang native __bf16 cast), 4 elems/thread
// ---------------------------------------------------------------------------
__global__ void cvt_f32_bf16(const float* __restrict__ in,
                             __bf16* __restrict__ out, int n) {
    int i = (blockIdx.x * blockDim.x + threadIdx.x) * 4;
    int stride = gridDim.x * blockDim.x * 4;
    for (; i < n; i += stride) {
        float4 v = *(const float4*)(in + i);
        v4bf o;
        o[0] = (__bf16)v.x; o[1] = (__bf16)v.y;
        o[2] = (__bf16)v.z; o[3] = (__bf16)v.w;
        *(v4bf*)(out + i) = o;
    }
}

// ---------------------------------------------------------------------------
// zero-fill (features buffer)
// ---------------------------------------------------------------------------
__global__ void zero_f32(float* __restrict__ p, size_t n) {
    size_t i = ((size_t)blockIdx.x * blockDim.x + threadIdx.x) * 4;
    size_t stride = (size_t)gridDim.x * blockDim.x * 4;
    float4 z = make_float4(0.f, 0.f, 0.f, 0.f);
    for (; i < n; i += stride) *(float4*)(p + i) = z;
}

// ---------------------------------------------------------------------------
// Encoder GEMM: pre[n,f] = sum_d x[n,d]*We[f,d] + be[f]
// bf16 WMMA, fp32 accumulate.  Block = 256 threads = 8 waves.
// Block tile 64(M) x 256(N); wave tile 32 x 64 (2x4 frags of 16x16).
// Double-buffered LDS staging via global_load_async_to_lds_b128 (ASYNCcnt),
// so the stage-(i+1) copy overlaps the stage-i WMMAs and the block tile is
// fetched from L2 exactly once (~51 FLOP / L2 byte).
// ---------------------------------------------------------------------------
__global__ __launch_bounds__(256)
void encoder_gemm(const __bf16* __restrict__ xb,
                  const __bf16* __restrict__ web,
                  const float*  __restrict__ be,
                  float* __restrict__ pre) {
    __shared__ __align__(16) __bf16 sA[2][64 * 32];    //  8 KB
    __shared__ __align__(16) __bf16 sB[2][256 * 32];   // 32 KB

    const int tid  = threadIdx.x;
    const int lane = tid & 31;
    const int wave = tid >> 5;             // 0..7
    const int wm   = wave & 1;             // 0..1
    const int wn   = wave >> 1;            // 0..3
    const int r    = lane & 15;
    const int hs   = lane >> 4;            // K-half select (wave32 layout)

    const int mBase0 = blockIdx.y * 64;
    const int nBase0 = blockIdx.x * 256;
    const int mBase  = mBase0 + wm * 32;
    const int nBase  = nBase0 + wn * 64;

    // copy mapping: A: thread t -> row t>>2 (64 rows), 16B segment t&3 (of 4)
    //               B: thread t -> col t (256 cols), 4 x 16B segments
    const int aRow = tid >> 2;
    const int aSeg = tid & 3;
    const __bf16* aGlob = xb  + (size_t)(mBase0 + aRow) * D_MODEL + aSeg * 8;
    const __bf16* bGlob = web + (size_t)(nBase0 + tid) * D_MODEL;
    const unsigned aLds0 = (unsigned)(uintptr_t)&sA[0][aRow * 32 + aSeg * 8];
    const unsigned aLds1 = (unsigned)(uintptr_t)&sA[1][aRow * 32 + aSeg * 8];
    const unsigned bLds0 = (unsigned)(uintptr_t)&sB[0][tid * 32];
    const unsigned bLds1 = (unsigned)(uintptr_t)&sB[1][tid * 32];

    v8f c[2][4];
#pragma unroll
    for (int mi = 0; mi < 2; ++mi)
#pragma unroll
        for (int ni = 0; ni < 4; ++ni)
            c[mi][ni] = (v8f){0.f, 0.f, 0.f, 0.f, 0.f, 0.f, 0.f, 0.f};

    // prologue: stage 0 into buffer 0 (5 async b128 per thread)
    async_b128(aLds0, aGlob);
#pragma unroll
    for (int s = 0; s < 4; ++s)
        async_b128(bLds0 + s * 16, bGlob + s * 8);

    for (int kk = 0; kk < D_MODEL; kk += 32) {
        const int buf = (kk >> 5) & 1;

        wait_async0();         // this wave's stage-i copies landed
        __syncthreads();       // all waves' copies landed; prev buffer free

        if (kk + 32 < D_MODEL) {
            const int knext = kk + 32;
            const unsigned aL = buf ? aLds0 : aLds1;
            const unsigned bL = buf ? bLds0 : bLds1;
            async_b128(aL, aGlob + knext);
#pragma unroll
            for (int s = 0; s < 4; ++s)
                async_b128(bL + s * 16, bGlob + knext + s * 8);
        }

        v16bf a[2], b[4];
        // A 16x32 bf16: hs==0 -> K{0..7,16..23}; hs==1 -> K{8..15,24..31}
#pragma unroll
        for (int mi = 0; mi < 2; ++mi) {
            const __bf16* p = &sA[buf][(wm * 32 + mi * 16 + r) * 32];
            v8bf lo = *(const v8bf*)(p + hs * 8);
            v8bf hi = *(const v8bf*)(p + 16 + hs * 8);
            a[mi] = __builtin_shufflevector(lo, hi, 0, 1, 2, 3, 4, 5, 6, 7,
                                            8, 9, 10, 11, 12, 13, 14, 15);
        }
        // B 32x16 bf16: hs==0 -> K{0..7,16..23}; hs==1 -> K{8..15,24..31}
#pragma unroll
        for (int ni = 0; ni < 4; ++ni) {
            const __bf16* p = &sB[buf][(wn * 64 + ni * 16 + r) * 32];
            v8bf lo = *(const v8bf*)(p + hs * 8);
            v8bf hi = *(const v8bf*)(p + 16 + hs * 8);
            b[ni] = __builtin_shufflevector(lo, hi, 0, 1, 2, 3, 4, 5, 6, 7,
                                            8, 9, 10, 11, 12, 13, 14, 15);
        }
#pragma unroll
        for (int mi = 0; mi < 2; ++mi)
#pragma unroll
            for (int ni = 0; ni < 4; ++ni)
                c[mi][ni] = __builtin_amdgcn_wmma_f32_16x16x32_bf16(
                    false, a[mi], false, b[ni], (short)0, c[mi][ni],
                    false, false);
    }

    // C/D layout: vgpr v -> M = v + 8*hs, N = lane&15
#pragma unroll
    for (int ni = 0; ni < 4; ++ni) {
        const int col = nBase + ni * 16 + r;
        const float bias = be[col];
#pragma unroll
        for (int mi = 0; mi < 2; ++mi) {
            const int row0 = mBase + mi * 16 + hs * 8;
#pragma unroll
            for (int v = 0; v < 8; ++v)
                pre[(size_t)(row0 + v) * DICT + col] = c[mi][ni][v] + bias;
        }
    }
}

// ---------------------------------------------------------------------------
// Top-K(32) per row over LDS-resident row (128KB of the 320KB WGP LDS).
// 32 argmax+knockout passes with a block tree reduction.
// Scatters relu(val) into dense features; records (idx,val) for the decoder.
// ---------------------------------------------------------------------------
__global__ __launch_bounds__(256)
void topk_kernel(const float* __restrict__ pre,
                 float* __restrict__ feat,
                 int*   __restrict__ tidx,
                 float* __restrict__ tval) {
    extern __shared__ unsigned char smem[];
    float* srow = (float*)smem;                                  // DICT floats
    float* redv = (float*)(smem + (size_t)DICT * 4);             // 256 floats
    int*   redi = (int*)  (smem + (size_t)DICT * 4 + 256 * 4);   // 256 ints

    const int n = blockIdx.x;
    const int tid = threadIdx.x;
    const float* row = pre + (size_t)n * DICT;

    for (int i = tid * 4; i < DICT; i += 256 * 4)
        *(float4*)(srow + i) = *(const float4*)(row + i);
    __syncthreads();

    const float NEG_INF = -__builtin_inff();
    for (int k = 0; k < TOPK; ++k) {
        float lv = NEG_INF;
        int li = 0;
        for (int i = tid; i < DICT; i += 256) {
            float v = srow[i];
            if (v > lv) { lv = v; li = i; }
        }
        redv[tid] = lv;
        redi[tid] = li;
        __syncthreads();
        for (int s = 128; s > 0; s >>= 1) {
            if (tid < s) {
                float ov = redv[tid + s];
                int   oi = redi[tid + s];
                if (ov > redv[tid] || (ov == redv[tid] && oi < redi[tid])) {
                    redv[tid] = ov;
                    redi[tid] = oi;
                }
            }
            __syncthreads();
        }
        if (tid == 0) {
            int bi = redi[0];
            float bv = redv[0];
            srow[bi] = NEG_INF;                       // knock out winner
            tidx[(size_t)n * TOPK + k] = bi;
            tval[(size_t)n * TOPK + k] = bv;
            feat[(size_t)n * DICT + bi] = fmaxf(bv, 0.f);
        }
        __syncthreads();
    }
}

// ---------------------------------------------------------------------------
// Sparse decoder: recon[n,d] = sum_j relu(val_j) * Wd[d, idx_j] + bd[d]
// One block per row; (idx,val) broadcast via LDS; each thread owns 8 d's.
// ---------------------------------------------------------------------------
__global__ __launch_bounds__(256)
void decoder_kernel(const int*   __restrict__ tidx,
                    const float* __restrict__ tval,
                    const float* __restrict__ Wd,
                    const float* __restrict__ bd,
                    float* __restrict__ recon) {
    const int n = blockIdx.x;
    const int tid = threadIdx.x;
    __shared__ int   sidx[TOPK];
    __shared__ float sval[TOPK];
    if (tid < TOPK) {
        sidx[tid] = tidx[(size_t)n * TOPK + tid];
        sval[tid] = fmaxf(tval[(size_t)n * TOPK + tid], 0.f);
    }
    __syncthreads();

    float acc[8];
#pragma unroll
    for (int i = 0; i < 8; ++i) acc[i] = bd[tid + 256 * i];

    for (int j = 0; j < TOPK; ++j) {
        const float v = sval[j];           // uniform across block
        if (v == 0.f) continue;
        const int f = sidx[j];
#pragma unroll
        for (int i = 0; i < 8; ++i)
            acc[i] += v * Wd[(size_t)(tid + 256 * i) * DICT + f];
    }
#pragma unroll
    for (int i = 0; i < 8; ++i)
        recon[(size_t)n * D_MODEL + tid + 256 * i] = acc[i];
}

// ---------------------------------------------------------------------------
extern "C" void kernel_launch(void* const* d_in, const int* in_sizes, int n_in,
                              void* d_out, int out_size, void* d_ws,
                              size_t ws_size, hipStream_t stream) {
    const float* x  = (const float*)d_in[0];
    const float* We = (const float*)d_in[1];
    const float* be = (const float*)d_in[2];
    const float* Wd = (const float*)d_in[3];
    const float* bd = (const float*)d_in[4];

    float* recon = (float*)d_out;                           // [N, D]
    float* feat  = recon + (size_t)NROWS * D_MODEL;         // [N, DICT]
    float* pre   = feat  + (size_t)NROWS * DICT;            // [N, DICT]

    // workspace: bf16 We (128MB) | bf16 x (8MB) | topk idx/val (512KB)
    __bf16* web = (__bf16*)d_ws;
    __bf16* xb  = web + (size_t)DICT * D_MODEL;
    int*    tidx = (int*)(xb + (size_t)NROWS * D_MODEL);
    float*  tval = (float*)(tidx + NROWS * TOPK);

    cvt_f32_bf16<<<2048, 256, 0, stream>>>(We, web, DICT * D_MODEL);
    cvt_f32_bf16<<<512, 256, 0, stream>>>(x, xb, NROWS * D_MODEL);
    zero_f32<<<2048, 256, 0, stream>>>(feat, (size_t)NROWS * DICT);

    dim3 grid(DICT / 256, NROWS / 64);  // 128 x 32 blocks
    encoder_gemm<<<grid, 256, 0, stream>>>(xb, web, be, pre);

    size_t smem = (size_t)DICT * 4 + 256 * 4 + 256 * 4;  // 130 KB < 320 KB WGP LDS
    topk_kernel<<<NROWS, 256, smem, stream>>>(pre, feat, tidx, tval);

    decoder_kernel<<<NROWS, 256, 0, stream>>>(tidx, tval, Wd, bd, recon);
}
